// MultiScaleRetention_65687229825765
// MI455X (gfx1250) — compile-verified, matrix operand order checked by reading, not run on hardware
//
#include <hip/hip_runtime.h>
#include <hip/hip_bf16.h>
#include <cstdint>

#define D_MODEL  1024
#define N_HEADS  16
#define HEAD_DIM 64
#define BATCH    2
#define SEQ      2048
#define M_TOTAL  (BATCH * SEQ)   // 4096
#define LDSS     72              // LDS tile row stride in halves (144B, 16B-aligned)

typedef __attribute__((ext_vector_type(16))) _Float16 v16h;
typedef __attribute__((ext_vector_type(8)))  _Float16 v8h;
typedef __attribute__((ext_vector_type(8)))  float    v8f;

// ---------------------------------------------------------------------------
// WMMA helpers (CDNA5 wave32, V_WMMA_F32_16X16X32_F16)
// 16-bit A-frag layout: lane m = lane&15, kbase = (lane>>4)*8,
//   VGPRs 0-3 = K kbase..kbase+7 (contiguous), VGPRs 4-7 = K 16+kbase..+7.
// => fragment = two 16-byte LDS loads per lane from a row-major tile whose
//    row stride is a multiple of 8 halves.  B-frag is identical with lane->N,
//    so B tiles are kept N-major (n x k) in LDS.
// C/D (16x16 f32, 8 VGPRs): M = r + 8*(lane>>4), N = lane&15.
// ---------------------------------------------------------------------------

__device__ __forceinline__ v8f vzero8() {
  v8f z;
#pragma unroll
  for (int i = 0; i < 8; ++i) z[i] = 0.0f;
  return z;
}

// p points at element (row0, k0) of a row-major f16 LDS tile (stride mult. of 8)
__device__ __forceinline__ v16h load_frag_row(const _Float16* p, int stride) {
  const int lane = threadIdx.x & 31;
  const _Float16* q = p + (lane & 15) * stride + ((lane >> 4) << 3);
  const v8h lo = *(const v8h*)q;          // K = kbase .. kbase+7
  const v8h hi = *(const v8h*)(q + 16);   // K = 16+kbase .. 16+kbase+7
  return __builtin_shufflevector(lo, hi, 0, 1, 2, 3, 4, 5, 6, 7,
                                         8, 9, 10, 11, 12, 13, 14, 15);
}

__device__ __forceinline__ v8f wmma16(v16h a, v16h b, v8f c) {
  return __builtin_amdgcn_wmma_f32_16x16x32_f16(false, a, false, b,
                                                (short)0, c, false, false);
}

// ---------------------------------------------------------------------------
// CDNA5 async global->LDS copy (GLOBAL_LOAD_ASYNC_TO_LDS_B128, ASYNCcnt).
// LDS byte offset = low 32 bits of the flat address.
// Each lane copies 16 bytes; wave waits its own ASYNCcnt before the barrier.
// ---------------------------------------------------------------------------
__device__ __forceinline__ void async_copy_b128(void* lds, const void* gptr) {
  const uint32_t lds_off = (uint32_t)(uintptr_t)lds;
  asm volatile("global_load_async_to_lds_b128 %0, %1, off"
               :
               : "v"(lds_off), "v"(gptr)
               : "memory");
}

__device__ __forceinline__ void wait_async() {
  asm volatile("s_wait_asynccnt 0x0" ::: "memory");
}

// ---------------------------------------------------------------------------
// Stage 0a: fp32 -> fp16 elementwise (for x)
// ---------------------------------------------------------------------------
__global__ __launch_bounds__(256) void cvt_f32_to_f16(const float* __restrict__ src,
                                                      _Float16* __restrict__ dst,
                                                      int n) {
  int i = blockIdx.x * 256 + threadIdx.x;
  if (i < n) dst[i] = (_Float16)src[i];
}

// ---------------------------------------------------------------------------
// Stage 0b: fp32 (K x N) -> fp16 transposed (N x K), tiled through LDS.
// blockIdx.x = k-tile, blockIdx.y = n-tile, blockIdx.z = matrix index.
// ---------------------------------------------------------------------------
__global__ __launch_bounds__(128) void transpose_f32_to_f16(
    const float* __restrict__ src, _Float16* __restrict__ dst, int K, int N) {
  __shared__ __align__(16) _Float16 t[64 * LDSS];
  const int tid = threadIdx.x;
  const size_t moff = (size_t)blockIdx.z * K * N;
  const int k0 = blockIdx.x * 64, n0 = blockIdx.y * 64;
#pragma unroll
  for (int ii = 0; ii < 4; ++ii) {
    const int idx = tid + ii * 128;
    const int r = idx >> 3, c8 = (idx & 7) << 3;
    const float* s = src + moff + (size_t)(k0 + r) * N + n0 + c8;
    v8h v;
#pragma unroll
    for (int i = 0; i < 8; ++i) v[i] = (_Float16)s[i];
    *(v8h*)&t[r * LDSS + c8] = v;
  }
  __syncthreads();
#pragma unroll
  for (int ii = 0; ii < 4; ++ii) {
    const int idx = tid + ii * 128;
    const int n = idx >> 3, k8 = (idx & 7) << 3;
    v8h v;
#pragma unroll
    for (int i = 0; i < 8; ++i) v[i] = t[(k8 + i) * LDSS + n];
    *(v8h*)(dst + moff + (size_t)(n0 + n) * K + k0 + k8) = v;
  }
}

// ---------------------------------------------------------------------------
// Stage 1: QKV projection.  128x64 block tile, 4 waves x 32 M-rows each.
// blockIdx.x = M tile (128 rows), blockIdx.y = mi*16+h
// A = xh (4096 x 1024) row-major, B = wT[(mi,h)] (64 x 1024) N-major.
// Output qkvh layout: [mi][b][h][l][e] f16
// ---------------------------------------------------------------------------
__global__ __launch_bounds__(128) void qkv_gemm(const _Float16* __restrict__ xh,
                                                const _Float16* __restrict__ wT,
                                                _Float16* __restrict__ qkvh) {
  __shared__ __align__(16) _Float16 At[128 * LDSS];
  __shared__ __align__(16) _Float16 Bt[64 * LDSS];
  const int mt   = blockIdx.x;                 // 0..31
  const int mh   = blockIdx.y;                 // mi*16 + h
  const int tid  = threadIdx.x;
  const int wv   = tid >> 5;
  const int lane = tid & 31;
  const _Float16* Wp = wT + (size_t)mh * (1024 * 64);   // [n][k]

  v8f acc[2][4];
#pragma unroll
  for (int s = 0; s < 2; ++s)
#pragma unroll
    for (int i = 0; i < 4; ++i) acc[s][i] = vzero8();

  for (int k0 = 0; k0 < 1024; k0 += 64) {
    __syncthreads();
#pragma unroll
    for (int ii = 0; ii < 8; ++ii) {               // At: 128x64 halves
      const int idx = tid + ii * 128;
      const int r = idx >> 3, c8 = (idx & 7) << 3;
      async_copy_b128(&At[r * LDSS + c8],
                      xh + (size_t)(mt * 128 + r) * 1024 + k0 + c8);
    }
#pragma unroll
    for (int ii = 0; ii < 4; ++ii) {               // Bt: 64x64 halves
      const int idx = tid + ii * 128;
      const int r = idx >> 3, c8 = (idx & 7) << 3;
      async_copy_b128(&Bt[r * LDSS + c8],
                      Wp + (size_t)r * 1024 + k0 + c8);
    }
    wait_async();
    __syncthreads();

#pragma unroll
    for (int ks = 0; ks < 64; ks += 32) {
      const v16h a0 = load_frag_row(At + (wv * 32 +  0) * LDSS + ks, LDSS);
      const v16h a1 = load_frag_row(At + (wv * 32 + 16) * LDSS + ks, LDSS);
#pragma unroll
      for (int nt = 0; nt < 4; ++nt) {
        const v16h bb = load_frag_row(Bt + nt * 16 * LDSS + ks, LDSS);
        acc[0][nt] = wmma16(a0, bb, acc[0][nt]);   // B frag feeds 2 WMMAs
        acc[1][nt] = wmma16(a1, bb, acc[1][nt]);
      }
    }
  }

#pragma unroll
  for (int s = 0; s < 2; ++s) {
    const int rowbase = mt * 128 + wv * 32 + s * 16 + ((lane >> 4) << 3);
#pragma unroll
    for (int nt = 0; nt < 4; ++nt) {
      const int e = nt * 16 + (lane & 15);
#pragma unroll
      for (int r = 0; r < 8; ++r) {
        const int row = rowbase + r;
        const int b   = row >> 11;          // / SEQ
        const int l   = row & 2047;
        const size_t base = (size_t)(mh >> 4) * (BATCH * N_HEADS)
                          + (size_t)b * N_HEADS + (mh & 15);
        qkvh[(base * SEQ + l) * HEAD_DIM + e] = (_Float16)acc[s][nt][r];
      }
    }
  }
}

// ---------------------------------------------------------------------------
// Stage 1b: transpose V (per (b,h): L x hd -> hd x L), tiled through LDS.
// blockIdx.x = l-tile, blockIdx.y = b*16+h
// ---------------------------------------------------------------------------
__global__ __launch_bounds__(128) void transpose_v(const _Float16* __restrict__ v,
                                                   _Float16* __restrict__ vt) {
  __shared__ __align__(16) _Float16 t[64 * LDSS];
  const int lt = blockIdx.x, bh = blockIdx.y;
  const int tid = threadIdx.x;
  const _Float16* vb = v + (size_t)bh * SEQ * HEAD_DIM;
  _Float16* vtb = vt + (size_t)bh * HEAD_DIM * SEQ;
#pragma unroll
  for (int ii = 0; ii < 4; ++ii) {
    const int idx = tid + ii * 128;
    const int r = idx >> 3, c8 = (idx & 7) << 3;
    *(v8h*)&t[r * LDSS + c8] =
        *(const v8h*)(vb + (size_t)(lt * 64 + r) * HEAD_DIM + c8);
  }
  __syncthreads();
#pragma unroll
  for (int ii = 0; ii < 4; ++ii) {
    const int idx = tid + ii * 128;
    const int e = idx >> 3, l8 = (idx & 7) << 3;
    v8h x;
#pragma unroll
    for (int i = 0; i < 8; ++i) x[i] = t[(l8 + i) * LDSS + e];
    *(v8h*)(vtb + (size_t)e * SEQ + lt * 64 + l8) = x;
  }
}

// ---------------------------------------------------------------------------
// Stage 2: retention.  blockIdx = (i-tile, head, batch).  Flash-style causal
// loop over j-tiles.  Q fragments are hoisted out of the jt loop; the decay
// mask uses precomputed per-lane powers (gamma^(i-j) = base*gi[r]*gj[nt]).
// ---------------------------------------------------------------------------
__global__ __launch_bounds__(128) void retention(const _Float16* __restrict__ qkvh,
                                                 const _Float16* __restrict__ vT,
                                                 float* __restrict__ ocomb) {
  const size_t QSZ = (size_t)BATCH * N_HEADS * SEQ * HEAD_DIM;
  __shared__ __align__(16) _Float16 Qt[64 * LDSS];
  __shared__ __align__(16) _Float16 Kt[64 * LDSS];
  __shared__ __align__(16) _Float16 Vt[64 * LDSS];   // e-major: [e][j]
  __shared__ __align__(16) _Float16 St[64 * LDSS];

  const int it   = blockIdx.x;   // 0..31
  const int h    = blockIdx.y;   // 0..15
  const int b    = blockIdx.z;   // 0..1
  const int tid  = threadIdx.x;
  const int wv   = tid >> 5;
  const int lane = tid & 31;

  // gamma_h = 1 - 2^-(5 + h/2)
  const float log2g = log2f(1.0f - exp2f(-(float)(5 + (h >> 1))));

  const _Float16* Qb  = qkvh + ((size_t)(b * N_HEADS + h) * SEQ) * HEAD_DIM;
  const _Float16* Kb  = Qb + QSZ;
  const _Float16* VTb = vT + ((size_t)(b * N_HEADS + h) * HEAD_DIM) * SEQ;

#pragma unroll
  for (int ii = 0; ii < 4; ++ii) {
    const int idx = tid + ii * 128;
    const int r = idx >> 3, c8 = (idx & 7) << 3;
    async_copy_b128(&Qt[r * LDSS + c8],
                    Qb + (size_t)(it * 64 + r) * HEAD_DIM + c8);
  }
  wait_async();
  __syncthreads();

  const int isub  = wv * 16 + ((lane >> 4) << 3);   // row within 64-tile
  const int ibase = it * 64 + isub;

  // jt-invariant Q fragments
  const v16h aq0 = load_frag_row(Qt + wv * 16 * LDSS + 0,  LDSS);
  const v16h aq1 = load_frag_row(Qt + wv * 16 * LDSS + 32, LDSS);

  // per-lane decay powers: gamma^(i-j) = base(jt) * gi[r] * gj[nt]
  float gi[8], gj[4];
#pragma unroll
  for (int r = 0; r < 8; ++r)
    gi[r] = exp2f(log2g * (float)(isub + r));
#pragma unroll
  for (int nt = 0; nt < 4; ++nt)
    gj[nt] = exp2f(-log2g * (float)(nt * 16 + (lane & 15)));

  v8f oacc[4];
#pragma unroll
  for (int i = 0; i < 4; ++i) oacc[i] = vzero8();

  for (int jt = 0; jt <= it; ++jt) {
    __syncthreads();
#pragma unroll
    for (int ii = 0; ii < 4; ++ii) {
      const int idx = tid + ii * 128;
      const int r = idx >> 3, c8 = (idx & 7) << 3;
      async_copy_b128(&Kt[r * LDSS + c8],
                      Kb + (size_t)(jt * 64 + r) * HEAD_DIM + c8);
      async_copy_b128(&Vt[r * LDSS + c8],
                      VTb + (size_t)r * SEQ + jt * 64 + c8);
    }
    wait_async();
    __syncthreads();

    // S = Q * K^T  (B-frag of K^T == A-pattern load of K)
    v8f sacc[4];
#pragma unroll
    for (int i = 0; i < 4; ++i) sacc[i] = vzero8();
#pragma unroll
    for (int nt = 0; nt < 4; ++nt) {
      const v16h bk0 = load_frag_row(Kt + nt * 16 * LDSS + 0,  LDSS);
      sacc[nt] = wmma16(aq0, bk0, sacc[nt]);
      const v16h bk1 = load_frag_row(Kt + nt * 16 * LDSS + 32, LDSS);
      sacc[nt] = wmma16(aq1, bk1, sacc[nt]);
    }

    // decay mask + 1/sqrt(hd); write own 16 rows of St (producer == consumer,
    // same-wave DS ordering keeps the RAW safe without a barrier)
    const float base = 0.125f * exp2f(log2g * (float)((it - jt) << 6));
#pragma unroll
    for (int nt = 0; nt < 4; ++nt) {
      const int jloc = nt * 16 + (lane & 15);
      const float bj = base * gj[nt];
#pragma unroll
      for (int r = 0; r < 8; ++r) {
        const int d = ((it - jt) << 6) + isub + r - jloc;
        const float wgt = (d >= 0) ? bj * gi[r] : 0.0f;
        St[(isub + r) * LDSS + jloc] = (_Float16)(sacc[nt][r] * wgt);
      }
    }

    // O += S * V
    const v16h as0 = load_frag_row(St + wv * 16 * LDSS + 0,  LDSS);
    const v16h as1 = load_frag_row(St + wv * 16 * LDSS + 32, LDSS);
#pragma unroll
    for (int nt = 0; nt < 4; ++nt) {
      const v16h bv0 = load_frag_row(Vt + nt * 16 * LDSS + 0,  LDSS);
      oacc[nt] = wmma16(as0, bv0, oacc[nt]);
      const v16h bv1 = load_frag_row(Vt + nt * 16 * LDSS + 32, LDSS);
      oacc[nt] = wmma16(as1, bv1, oacc[nt]);
    }
  }

#pragma unroll
  for (int nt = 0; nt < 4; ++nt) {
    const int e = nt * 16 + (lane & 15);
#pragma unroll
    for (int r = 0; r < 8; ++r) {
      const int l = ibase + r;
      ocomb[((size_t)b * SEQ + l) * D_MODEL + h * HEAD_DIM + e] = oacc[nt][r];
    }
  }
}

// ---------------------------------------------------------------------------
// Stage 3: GroupNorm stats per (b, head) over (head_dim, L)
// ---------------------------------------------------------------------------
__global__ __launch_bounds__(256) void gn_stats(const float* __restrict__ ocomb,
                                                float* __restrict__ stats) {
  __shared__ float ssum[256];
  __shared__ float ssq[256];
  const int bh = blockIdx.x;
  const int b = bh >> 4, h = bh & 15;
  float s = 0.0f, q = 0.0f;
  for (int idx = threadIdx.x; idx < SEQ * HEAD_DIM; idx += 256) {
    const int l = idx >> 6, e = idx & 63;
    const float v = ocomb[((size_t)b * SEQ + l) * D_MODEL + h * HEAD_DIM + e];
    s += v; q += v * v;
  }
  ssum[threadIdx.x] = s;
  ssq[threadIdx.x]  = q;
  __syncthreads();
  for (int off = 128; off > 0; off >>= 1) {
    if ((int)threadIdx.x < off) {
      ssum[threadIdx.x] += ssum[threadIdx.x + off];
      ssq[threadIdx.x]  += ssq[threadIdx.x + off];
    }
    __syncthreads();
  }
  if (threadIdx.x == 0) {
    const float n = (float)(SEQ * HEAD_DIM);
    const float mean = ssum[0] / n;
    const float var  = ssq[0] / n - mean * mean;
    stats[bh * 2]     = mean;
    stats[bh * 2 + 1] = rsqrtf(var + 1e-5f);
  }
}

// ---------------------------------------------------------------------------
// Stage 4: normalize + affine, emit f16 activations for the output GEMM
// ---------------------------------------------------------------------------
__global__ __launch_bounds__(256) void gn_apply(const float* __restrict__ ocomb,
                                                const float* __restrict__ stats,
                                                const float* __restrict__ gw,
                                                const float* __restrict__ gb,
                                                _Float16* __restrict__ yh) {
  const size_t i = (size_t)blockIdx.x * 256 + threadIdx.x;   // exact 4096*1024
  const int c  = (int)(i & 1023);
  const int bl = (int)(i >> 10);
  const int b  = bl >> 11;
  const int h  = c >> 6;
  const float m  = stats[(b * 16 + h) * 2];
  const float rs = stats[(b * 16 + h) * 2 + 1];
  const float v  = (ocomb[i] - m) * rs * gw[c] + gb[c];
  yh[i] = (_Float16)v;
}

// ---------------------------------------------------------------------------
// Stage 5: out projection.  128x64 block tile, 4 waves x 32 M-rows each.
// A = yh (4096x1024), B = woutT (1024x1024, N-major)
// blockIdx.x = M tile (128 rows), blockIdx.y = 64-col group
// ---------------------------------------------------------------------------
__global__ __launch_bounds__(128) void out_gemm(const _Float16* __restrict__ yh,
                                                const _Float16* __restrict__ wT,
                                                float* __restrict__ out) {
  __shared__ __align__(16) _Float16 At[128 * LDSS];
  __shared__ __align__(16) _Float16 Bt[64 * LDSS];
  const int mt   = blockIdx.x;
  const int ng   = blockIdx.y;
  const int tid  = threadIdx.x;
  const int wv   = tid >> 5;
  const int lane = tid & 31;

  v8f acc[2][4];
#pragma unroll
  for (int s = 0; s < 2; ++s)
#pragma unroll
    for (int i = 0; i < 4; ++i) acc[s][i] = vzero8();

  for (int k0 = 0; k0 < 1024; k0 += 64) {
    __syncthreads();
#pragma unroll
    for (int ii = 0; ii < 8; ++ii) {
      const int idx = tid + ii * 128;
      const int r = idx >> 3, c8 = (idx & 7) << 3;
      async_copy_b128(&At[r * LDSS + c8],
                      yh + (size_t)(mt * 128 + r) * 1024 + k0 + c8);
    }
#pragma unroll
    for (int ii = 0; ii < 4; ++ii) {
      const int idx = tid + ii * 128;
      const int r = idx >> 3, c8 = (idx & 7) << 3;
      async_copy_b128(&Bt[r * LDSS + c8],
                      wT + (size_t)(ng * 64 + r) * 1024 + k0 + c8);
    }
    wait_async();
    __syncthreads();

#pragma unroll
    for (int ks = 0; ks < 64; ks += 32) {
      const v16h a0 = load_frag_row(At + (wv * 32 +  0) * LDSS + ks, LDSS);
      const v16h a1 = load_frag_row(At + (wv * 32 + 16) * LDSS + ks, LDSS);
#pragma unroll
      for (int nt = 0; nt < 4; ++nt) {
        const v16h bb = load_frag_row(Bt + nt * 16 * LDSS + ks, LDSS);
        acc[0][nt] = wmma16(a0, bb, acc[0][nt]);
        acc[1][nt] = wmma16(a1, bb, acc[1][nt]);
      }
    }
  }

#pragma unroll
  for (int s = 0; s < 2; ++s) {
    const int rowbase = mt * 128 + wv * 32 + s * 16 + ((lane >> 4) << 3);
#pragma unroll
    for (int nt = 0; nt < 4; ++nt) {
      const int col = ng * 64 + nt * 16 + (lane & 15);
#pragma unroll
      for (int r = 0; r < 8; ++r) {
        out[(size_t)(rowbase + r) * 1024 + col] = acc[s][nt][r];
      }
    }
  }
}

// ---------------------------------------------------------------------------
// Host launcher.  Workspace layout (MiB offsets):
//   [ 0,  8)  xh  (f16 x)        — reused as yh after stage 4
//   [ 8, 14)  wT  (f16, Wq|Wk|Wv transposed: [mi][h][64][1024])
//   [14, 16)  woutT (f16 Wout transposed: [1024 n][1024 k])
//   [16, 40)  qkvh (f16 Q|K|V: [mi][b][h][l][e])
//   [40, 48)  vT   (f16 V transposed: [b][h][e][l])
//   [48, 64)  ocomb (f32 retention output, (B, L, D))
//   [64, +256B) stats
// ---------------------------------------------------------------------------
extern "C" void kernel_launch(void* const* d_in, const int* in_sizes, int n_in,
                              void* d_out, int out_size, void* d_ws, size_t ws_size,
                              hipStream_t stream) {
  const float* x    = (const float*)d_in[0];
  const float* Wq   = (const float*)d_in[1];
  const float* Wk   = (const float*)d_in[2];
  const float* Wv   = (const float*)d_in[3];
  const float* gw   = (const float*)d_in[4];
  const float* gb   = (const float*)d_in[5];
  const float* Wout = (const float*)d_in[6];
  float* out = (float*)d_out;

  char* ws = (char*)d_ws;
  _Float16* xh    = (_Float16*)(ws);
  _Float16* wT    = (_Float16*)(ws + ((size_t)8  << 20));
  _Float16* woutT = (_Float16*)(ws + ((size_t)14 << 20));
  _Float16* qkvh  = (_Float16*)(ws + ((size_t)16 << 20));
  _Float16* vT    = (_Float16*)(ws + ((size_t)40 << 20));
  float*    ocomb = (float*)   (ws + ((size_t)48 << 20));
  float*    stats = (float*)   (ws + ((size_t)64 << 20));

  const int NX = M_TOTAL * D_MODEL;                 // 4194304
  const int NW = N_HEADS * D_MODEL * HEAD_DIM;      // 1048576
  const size_t QSZ = (size_t)BATCH * N_HEADS * SEQ * HEAD_DIM;

  cvt_f32_to_f16<<<NX / 256, 256, 0, stream>>>(x, xh, NX);
  // per-head (1024 x 64) -> (64 x 1024), 16 matrices each
  transpose_f32_to_f16<<<dim3(16, 1, 16), 128, 0, stream>>>(Wq, wT,          1024, 64);
  transpose_f32_to_f16<<<dim3(16, 1, 16), 128, 0, stream>>>(Wk, wT + NW,     1024, 64);
  transpose_f32_to_f16<<<dim3(16, 1, 16), 128, 0, stream>>>(Wv, wT + 2 * NW, 1024, 64);
  // Wout (1024 x 1024) -> (1024 x 1024) transposed
  transpose_f32_to_f16<<<dim3(16, 16, 1), 128, 0, stream>>>(Wout, woutT, 1024, 1024);

  qkv_gemm<<<dim3(M_TOTAL / 128, 48), 128, 0, stream>>>(xh, wT, qkvh);
  transpose_v<<<dim3(SEQ / 64, BATCH * N_HEADS), 128, 0, stream>>>(qkvh + 2 * QSZ, vT);
  retention<<<dim3(SEQ / 64, N_HEADS, BATCH), 128, 0, stream>>>(qkvh, vT, ocomb);
  gn_stats<<<BATCH * N_HEADS, 256, 0, stream>>>(ocomb, stats);
  gn_apply<<<NX / 256, 256, 0, stream>>>(ocomb, stats, gw, gb, xh /* -> yh */);
  out_gemm<<<dim3(M_TOTAL / 128, D_MODEL / 64), 128, 0, stream>>>(xh, woutT, out);
}
